// WSchnet_G_13443247637171
// MI455X (gfx1250) — matrix operand order repaired and checked by reading, not compiled, not run on hardware
//
#include <hip/hip_runtime.h>
#include <hip/hip_bf16.h>
#include <math.h>

#define NN 50000
#define NE 800000
#define NG 500
#define DIM 64
#define RBFD 5
#define NCONV 3
#define CLSD 2000
#define HID 256

typedef __bf16 bf16_t;
typedef __attribute__((ext_vector_type(16))) __bf16 v16bf;
typedef __attribute__((ext_vector_type(8)))  __bf16 v8bf;
typedef __attribute__((ext_vector_type(8)))  float  v8f;

union AFrag { v16bf v; v8bf h[2]; bf16_t e[16]; };

// activations: 0=none 1=sp2 (2*softplus(0.5x)) 2=softplus(x)-ln2 3=relu
__device__ __forceinline__ float act_apply(int act, float x) {
  switch (act) {
    case 1: { float h = 0.5f * x;
              return 2.0f * (fmaxf(h, 0.0f) + __logf(1.0f + __expf(-fabsf(h)))); }
    case 2: return fmaxf(x, 0.0f) + __logf(1.0f + __expf(-fabsf(x))) - 0.6931471805599453f;
    case 3: return fmaxf(x, 0.0f);
    default: return x;
  }
}

// ---------------------------------------------------------------------------
// Generic bf16 WMMA GEMM: out = act( A[M,K] @ W[K,N] + bias (+ resid) )
// A row-major (bf16, or f32 converted on load; compile-time choice), Wt is W
// transposed [N][K] bf16. 128 threads = 4 waves; each wave owns one 16x16
// output tile (4 consecutive col tiles per block). Loads are clamped (no
// exec-mask divergence in the K loop); only stores are guarded.
// ---------------------------------------------------------------------------
template <bool ABF>
__global__ void gemm_wmma(const void*   __restrict__ Aptr,
                          const bf16_t* __restrict__ Wt,
                          const float*  __restrict__ bias,
                          const float*  __restrict__ resid,
                          float*        __restrict__ outF,
                          bf16_t*       __restrict__ outB,
                          int M, int N, int K, int actF, int actB) {
  const int lane = threadIdx.x & 31;
  const int wave = threadIdx.x >> 5;
  const int hi   = lane >> 4;
  const int lr   = lane & 15;
  const int row0 = blockIdx.x * 16;
  const int col0 = (blockIdx.y * 4 + wave) * 16;

  const int aRow = min(row0 + lr, M - 1);          // clamped: always a valid load
  const int bCol = min(col0 + lr, N - 1);
  const size_t aBase = (size_t)aRow * K;
  const size_t bBase = (size_t)bCol * K;
  const bf16_t* Abf  = (const bf16_t*)Aptr;
  const float*  Af32 = (const float*)Aptr;

  v8f acc = {};
  for (int kk = 0; kk < K; kk += 32) {
    AFrag a, b;
    if (ABF) {
      const bf16_t* ap = Abf + aBase + kk + 8 * hi;
      a.h[0] = *(const v8bf*)(ap);        // K = kk+8*hi .. +7
      a.h[1] = *(const v8bf*)(ap + 16);   // K = kk+16+8*hi .. +7
    } else {
      const float* ap = Af32 + aBase + kk + 8 * hi;
#pragma unroll
      for (int j = 0; j < 8; ++j) {
        a.e[j]     = (bf16_t)ap[j];
        a.e[8 + j] = (bf16_t)ap[16 + j];
      }
    }
    const bf16_t* bp = Wt + bBase + kk + 8 * hi;  // K = kk+8*hi .. +7, kk+16+8*hi .. +7
    b.h[0] = *(const v8bf*)(bp);
    b.h[1] = *(const v8bf*)(bp + 16);
    acc = __builtin_amdgcn_wmma_f32_16x16x32_bf16(false, a.v, false, b.v,
                                                  (short)0, acc, false, false);
  }

  const int col = col0 + lr;
  float bval = bias ? bias[min(col, N - 1)] : 0.0f;
#pragma unroll
  for (int v = 0; v < 8; ++v) {
    int row = row0 + v + 8 * hi;
    if (row < M && col < N) {
      size_t idx = (size_t)row * N + col;
      float r = acc[v] + bval;
      if (resid) r += resid[idx];
      if (outF) outF[idx] = act_apply(actF, r);
      if (outB) outB[idx] = (bf16_t)act_apply(actB, r);
    }
  }
}

// ---------------------------------------------------------------------------
// Fused edge kernel: eh2 = sp2(eh1 @ cf2 + b); msg = eh2 * new_node[src];
// agg[dst] += msg (f32 global atomics). 4 waves/block; each wave handles
// 16 edges x 64 cols. Requires E % 16 == 0 (true: 800000) -> no lane guards.
// ---------------------------------------------------------------------------
__global__ void edge_conv(const bf16_t* __restrict__ eh1,      // [E,64]
                          const bf16_t* __restrict__ cf2t,     // [64][64] transposed
                          const float*  __restrict__ cf2b,     // [64]
                          const float*  __restrict__ new_node, // [NN,64]
                          const int* __restrict__ src, const int* __restrict__ dst,
                          float* __restrict__ agg, int E) {
  const int lane = threadIdx.x & 31;
  const int wave = threadIdx.x >> 5;
  const int hi = lane >> 4, lr = lane & 15;
  const int e0 = (blockIdx.x * 4 + wave) * 16;
  if (e0 >= E) return;                       // uniform per wave

  int sIdx[8], dIdx[8];
#pragma unroll
  for (int v = 0; v < 8; ++v) {
    int e = e0 + v + 8 * hi;                 // C-layout row owned by this lane
    sIdx[v] = src[e];
    dIdx[v] = dst[e];
  }

  const bf16_t* ap = eh1 + (size_t)(e0 + lr) * 64;
  AFrag a0, a1;
  a0.h[0] = *(const v8bf*)(ap + 8 * hi);
  a0.h[1] = *(const v8bf*)(ap + 16 + 8 * hi);
  a1.h[0] = *(const v8bf*)(ap + 32 + 8 * hi);
  a1.h[1] = *(const v8bf*)(ap + 48 + 8 * hi);

#pragma unroll
  for (int cn = 0; cn < 4; ++cn) {
    const int col = cn * 16 + lr;
    AFrag b0, b1;
    const bf16_t* bp = cf2t + (size_t)col * 64 + 8 * hi;
    b0.h[0] = *(const v8bf*)(bp);       b0.h[1] = *(const v8bf*)(bp + 16);
    b1.h[0] = *(const v8bf*)(bp + 32);  b1.h[1] = *(const v8bf*)(bp + 48);

    v8f acc = {};
    acc = __builtin_amdgcn_wmma_f32_16x16x32_bf16(false, a0.v, false, b0.v,
                                                  (short)0, acc, false, false);
    acc = __builtin_amdgcn_wmma_f32_16x16x32_bf16(false, a1.v, false, b1.v,
                                                  (short)0, acc, false, false);
    float bb = cf2b[col];
#pragma unroll
    for (int v = 0; v < 8; ++v) {
      float val = act_apply(1, acc[v] + bb);                 // sp2
      val *= new_node[(size_t)sIdx[v] * 64 + col];           // gather u_mul_e
      atomicAdd(&agg[(size_t)dIdx[v] * 64 + col], val);      // segment_sum
    }
  }
}

// ---------------------------------------------------------------------------
// Small helper kernels
// ---------------------------------------------------------------------------
__global__ void transpose_bf16(const float* __restrict__ src, bf16_t* __restrict__ dst,
                               int rows, int cols) {
  int idx = blockIdx.x * blockDim.x + threadIdx.x;
  if (idx < rows * cols) {
    int r = idx / cols, c = idx % cols;
    dst[(size_t)c * rows + r] = (bf16_t)src[idx];
  }
}

__global__ void zero_f32(float* p, long n) {
  long i = (long)blockIdx.x * blockDim.x + threadIdx.x;
  long stride = (long)gridDim.x * blockDim.x;
  for (; i < n; i += stride) p[i] = 0.0f;
}

__global__ void embed_kernel(const int* __restrict__ types, const float* __restrict__ emb,
                             float* __restrict__ nodeF, bf16_t* __restrict__ nodeB) {
  int idx = blockIdx.x * blockDim.x + threadIdx.x;
  if (idx < NN * DIM) {
    int n = idx >> 6, d = idx & 63;
    float v = emb[(size_t)types[n] * DIM + d];
    nodeF[idx] = v; nodeB[idx] = (bf16_t)v;
  }
}

__global__ void rbf_kernel(const float* __restrict__ dist, float* __restrict__ rbf) {
  int e = blockIdx.x * blockDim.x + threadIdx.x;
  if (e < NE) {
    float d = dist[e];
#pragma unroll
    for (int k = 0; k < RBFD; ++k) {
      float t = d - 1.25f * k;                 // centers = linspace(0,5,5), gap=1.25
      rbf[(size_t)e * RBFD + k] = __expf(-0.8f * t * t);   // 1/gap = 0.8
    }
  }
}

__global__ void eh1_kernel(const float* __restrict__ rbf, const float* __restrict__ cf1w,
                           const float* __restrict__ cf1b, bf16_t* __restrict__ eh1) {
  int idx = blockIdx.x * blockDim.x + threadIdx.x;
  if (idx < NE * DIM) {
    int e = idx >> 6, c = idx & 63;
    const float* r = rbf + (size_t)e * RBFD;
    float s = cf1b[c];
#pragma unroll
    for (int k = 0; k < RBFD; ++k) s += r[k] * cf1w[k * DIM + c];
    eh1[idx] = (bf16_t)act_apply(1, s);        // sp2
  }
}

__global__ void pool_accum(const float* __restrict__ res, const int* __restrict__ gid,
                           float* __restrict__ psum) {
  int idx = blockIdx.x * blockDim.x + threadIdx.x;
  if (idx < NN * HID) {
    int n = idx >> 8, c = idx & 255;
    atomicAdd(&psum[(size_t)gid[n] * HID + c], res[idx]);
  }
}

__global__ void count_kernel(const int* __restrict__ gid, float* __restrict__ cnt) {
  int n = blockIdx.x * blockDim.x + threadIdx.x;
  if (n < NN) atomicAdd(&cnt[gid[n]], 1.0f);
}

__global__ void pool_div(const float* __restrict__ psum, const float* __restrict__ cnt,
                         bf16_t* __restrict__ embg) {
  int idx = blockIdx.x * blockDim.x + threadIdx.x;
  if (idx < NG * HID) {
    int g = idx >> 8;
    embg[idx] = (bf16_t)(psum[idx] / fmaxf(cnt[g], 1.0f));
  }
}

// ---------------------------------------------------------------------------
static void launch_gemm(hipStream_t s, const void* A, bool aIsBf16,
                        const bf16_t* Wt, const float* bias, const float* resid,
                        float* outF, bf16_t* outB, int M, int N, int K,
                        int actF, int actB) {
  int nTiles = (N + 15) / 16;
  dim3 g((M + 15) / 16, (nTiles + 3) / 4);
  if (aIsBf16)
    gemm_wmma<true><<<g, 128, 0, s>>>(A, Wt, bias, resid, outF, outB, M, N, K, actF, actB);
  else
    gemm_wmma<false><<<g, 128, 0, s>>>(A, Wt, bias, resid, outF, outB, M, N, K, actF, actB);
}

extern "C" void kernel_launch(void* const* d_in, const int* in_sizes, int n_in,
                              void* d_out, int out_size, void* d_ws, size_t ws_size,
                              hipStream_t stream) {
  const int*   node_types = (const int*)d_in[0];
  const int*   edge_src   = (const int*)d_in[1];
  const int*   edge_dst   = (const int*)d_in[2];
  const int*   graph_ids  = (const int*)d_in[3];
  const float* distance   = (const float*)d_in[4];
  const float* emb        = (const float*)d_in[5];
  const float* conv_w1    = (const float*)d_in[6];
  const float* cf1_w      = (const float*)d_in[7];
  const float* cf1_b      = (const float*)d_in[8];
  const float* cf2_w      = (const float*)d_in[9];
  const float* cf2_b      = (const float*)d_in[10];
  const float* nl2_w      = (const float*)d_in[11];
  const float* nl2_b      = (const float*)d_in[12];
  const float* nl3_w      = (const float*)d_in[13];
  const float* nl3_b      = (const float*)d_in[14];
  const float* d1_w = (const float*)d_in[15];
  const float* d1_b = (const float*)d_in[16];
  const float* d2_w = (const float*)d_in[17];
  const float* d2_b = (const float*)d_in[18];
  const float* cls_w = (const float*)d_in[19];
  const float* cls_b = (const float*)d_in[20];
  const float* ac_w = (const float*)d_in[21];
  const float* ac_b = (const float*)d_in[22];

  // ---- workspace carve ----
  char* p = (char*)d_ws;
  auto take = [&](size_t bytes) -> void* {
    void* r = (void*)p; p += (bytes + 255) & ~(size_t)255; return r;
  };
  float*  nodeF   = (float*)take((size_t)NN * DIM * 4);
  bf16_t* nodeB   = (bf16_t*)take((size_t)NN * DIM * 2);
  float*  newnode = (float*)take((size_t)NN * DIM * 4);
  float*  agg     = (float*)take((size_t)NN * DIM * 4);
  bf16_t* tB      = (bf16_t*)take((size_t)NN * DIM * 2);
  float*  rbfBuf  = (float*)take((size_t)NE * RBFD * 4);
  bf16_t* eh1Buf  = (bf16_t*)take((size_t)NE * DIM * 2);
  bf16_t* atomB   = (bf16_t*)take((size_t)NN * HID * 2);
  float*  resF    = (float*)take((size_t)NN * HID * 4);
  bf16_t* reluB   = (bf16_t*)take((size_t)NN * HID * 2);
  float*  psum    = (float*)take((size_t)NG * HID * 4);
  float*  cnt     = (float*)take((size_t)NG * 4);
  bf16_t* embgB   = (bf16_t*)take((size_t)NG * HID * 2);
  bf16_t* w1t  = (bf16_t*)take((size_t)NCONV * DIM * DIM * 2);
  bf16_t* cf2t = (bf16_t*)take((size_t)NCONV * DIM * DIM * 2);
  bf16_t* nl2t = (bf16_t*)take((size_t)NCONV * DIM * DIM * 2);
  bf16_t* nl3t = (bf16_t*)take((size_t)NCONV * DIM * DIM * 2);
  bf16_t* d1t  = (bf16_t*)take((size_t)DIM * HID * 2);
  bf16_t* d2t  = (bf16_t*)take((size_t)HID * HID * 2);
  bf16_t* act_t = (bf16_t*)take((size_t)HID * 100 * 2);
  bf16_t* clst = (bf16_t*)take((size_t)HID * CLSD * 2);

  const int TB = 256;
  // ---- prep: transpose all weights to bf16 [out][in] ----
  for (int i = 0; i < NCONV; ++i) {
    int n = DIM * DIM, g = (n + TB - 1) / TB;
    transpose_bf16<<<g, TB, 0, stream>>>(conv_w1 + (size_t)i * n, w1t + (size_t)i * n, DIM, DIM);
    transpose_bf16<<<g, TB, 0, stream>>>(cf2_w   + (size_t)i * n, cf2t + (size_t)i * n, DIM, DIM);
    transpose_bf16<<<g, TB, 0, stream>>>(nl2_w   + (size_t)i * n, nl2t + (size_t)i * n, DIM, DIM);
    transpose_bf16<<<g, TB, 0, stream>>>(nl3_w   + (size_t)i * n, nl3t + (size_t)i * n, DIM, DIM);
  }
  transpose_bf16<<<(DIM * HID + TB - 1) / TB, TB, 0, stream>>>(d1_w, d1t, DIM, HID);
  transpose_bf16<<<(HID * HID + TB - 1) / TB, TB, 0, stream>>>(d2_w, d2t, HID, HID);
  transpose_bf16<<<(HID * 100 + TB - 1) / TB, TB, 0, stream>>>(ac_w, act_t, HID, 100);
  transpose_bf16<<<(HID * CLSD + TB - 1) / TB, TB, 0, stream>>>(cls_w, clst, HID, CLSD);

  // ---- embedding + RBF (layer independent) ----
  embed_kernel<<<(NN * DIM + TB - 1) / TB, TB, 0, stream>>>(node_types, emb, nodeF, nodeB);
  rbf_kernel<<<(NE + TB - 1) / TB, TB, 0, stream>>>(distance, rbfBuf);

  // ---- interaction layers ----
  for (int i = 0; i < NCONV; ++i) {
    const size_t wo = (size_t)i * DIM * DIM;
    // new_node = node @ w1
    launch_gemm(stream, nodeB, true, w1t + wo, nullptr, nullptr,
                newnode, nullptr, NN, DIM, DIM, 0, 0);
    // eh1 = sp2(rbf @ cf1 + b)
    eh1_kernel<<<(NE * DIM + TB - 1) / TB, TB, 0, stream>>>(
        rbfBuf, cf1_w + (size_t)i * RBFD * DIM, cf1_b + (size_t)i * DIM, eh1Buf);
    // agg = 0; fused edge GEMM + gather*mul + atomic scatter
    zero_f32<<<2048, TB, 0, stream>>>(agg, (long)NN * DIM);
    edge_conv<<<(NE + 63) / 64, 128, 0, stream>>>(
        eh1Buf, cf2t + wo, cf2_b + (size_t)i * DIM, newnode,
        edge_src, edge_dst, agg, NE);
    // t = sp2(agg @ nl2 + b)
    launch_gemm(stream, agg, false, nl2t + wo, nl2_b + (size_t)i * DIM, nullptr,
                nullptr, tB, NN, DIM, DIM, 0, 1);
    // node = node + t @ nl3 + b
    launch_gemm(stream, tB, true, nl3t + wo, nl3_b + (size_t)i * DIM, nodeF,
                nodeF, nodeB, NN, DIM, DIM, 0, 0);
  }

  // ---- dense head ----
  // atom = shiftsoftplus(node @ d1 + b)  (kept only as bf16 GEMM input)
  launch_gemm(stream, nodeB, true, d1t, d1_b, nullptr,
              nullptr, atomB, NN, HID, DIM, 0, 2);
  // res = atom @ d2 + b ; relu(res) as bf16 for atom classifier
  launch_gemm(stream, atomB, true, d2t, d2_b, nullptr,
              resF, reluB, NN, HID, HID, 0, 3);
  // atoms_preds = relu(res) @ ac + b  -> d_out[0 : NN*100)
  launch_gemm(stream, reluB, true, act_t, ac_b, nullptr,
              (float*)d_out, nullptr, NN, 100, HID, 0, 0);

  // ---- graph pooling + classifier ----
  zero_f32<<<512, TB, 0, stream>>>(psum, (long)NG * HID);
  zero_f32<<<4, TB, 0, stream>>>(cnt, (long)NG);
  pool_accum<<<(NN * HID + TB - 1) / TB, TB, 0, stream>>>(resF, graph_ids, psum);
  count_kernel<<<(NN + TB - 1) / TB, TB, 0, stream>>>(graph_ids, cnt);
  pool_div<<<(NG * HID + TB - 1) / TB, TB, 0, stream>>>(psum, cnt, embgB);
  // cls_preds = emb_g @ cls + b -> d_out[NN*100 : )
  launch_gemm(stream, embgB, true, clst, cls_b, nullptr,
              (float*)d_out + (size_t)NN * 100, nullptr, NG, CLSD, HID, 0, 0);
}
